// MaskedCrossAttention_22840636080944
// MI455X (gfx1250) — matrix operand order repaired and checked by reading, not compiled, not run on hardware
//
#include <hip/hip_runtime.h>

// ---------------------------------------------------------------------------
// MaskedCrossAttention for MI455X (gfx1250), bf16 WMMA pipeline.
//   x:(4,2048,2048) media:(4,8,64,1024) -> out:(4,2048,2048)
// ---------------------------------------------------------------------------

typedef __attribute__((ext_vector_type(16))) __bf16 v16bf;
typedef __attribute__((ext_vector_type(8)))  __bf16 v8bf;
typedef __attribute__((ext_vector_type(4)))  __bf16 v4bf;
typedef __attribute__((ext_vector_type(8)))  float  v8f;

#define BATCH    4
#define T_TXT    2048
#define DIMX     2048
#define T_IMG    8
#define N_LAT    64
#define DIM_VIS  1024
#define HEADS    8
#define DIM_HEAD 64
#define INNER    512      // HEADS * DIM_HEAD
#define NKEY     512      // T_IMG * N_LAT
#define M_TXT    (BATCH * T_TXT)   // 8192
#define M_KV     (BATCH * NKEY)    // 2048

// padded LDS row strides (20-bank lane stride -> conflict-free, 16B aligned)
#define ASTR 40
#define BSTR 40
#define VSTR 40
#define PSTR 40

static __device__ __forceinline__ v16bf cat8(v8bf lo, v8bf hi)
{
    return __builtin_shufflevector(lo, hi, 0, 1, 2, 3, 4, 5, 6, 7,
                                           8, 9, 10, 11, 12, 13, 14, 15);
}

// A-operand fragment (16x32 bf16): per lane two contiguous 8-element runs:
//   K = half*8 + [0,8)  and  K = 16 + half*8 + [0,8)
static __device__ __forceinline__ v16bf load_afrag(const __bf16* rowp, int half)
{
    const __bf16* p = rowp + (half << 3);
    return cat8(*(const v8bf*)p, *(const v8bf*)(p + 16));
}

// B-operand fragment from K-contiguous storage: per lane two contiguous 8-element runs:
//   K = half*8 + [0,8)  and  K = 16 + half*8 + [0,8)
static __device__ __forceinline__ v16bf load_bfrag(const __bf16* colp, int half)
{
    const __bf16* p = colp + (half << 3);
    return cat8(*(const v8bf*)p, *(const v8bf*)(p + 16));
}

// ---------------------------------------------------------------------------
// LayerNorm over last dim (2048) + convert to bf16. One block per row.
// ---------------------------------------------------------------------------
__global__ __launch_bounds__(256)
void mca_ln_kernel(const float* __restrict__ x, const float* __restrict__ w,
                   const float* __restrict__ bias, __bf16* __restrict__ xn)
{
    __shared__ float red[256];
    const int row = blockIdx.x;
    const float* xr = x + (size_t)row * DIMX;

    float s = 0.f;
    for (int c = threadIdx.x; c < DIMX; c += 256) s += xr[c];
    red[threadIdx.x] = s;
    __syncthreads();
    for (int off = 128; off > 0; off >>= 1) {
        if (threadIdx.x < off) red[threadIdx.x] += red[threadIdx.x + off];
        __syncthreads();
    }
    const float mu = red[0] * (1.0f / DIMX);
    __syncthreads();

    float s2 = 0.f;
    for (int c = threadIdx.x; c < DIMX; c += 256) {
        float d = xr[c] - mu;
        s2 += d * d;
    }
    red[threadIdx.x] = s2;
    __syncthreads();
    for (int off = 128; off > 0; off >>= 1) {
        if (threadIdx.x < off) red[threadIdx.x] += red[threadIdx.x + off];
        __syncthreads();
    }
    const float rstd = rsqrtf(red[0] * (1.0f / DIMX) + 1e-5f);

    __bf16* xo = xn + (size_t)row * DIMX;
    for (int c = threadIdx.x; c < DIMX; c += 256)
        xo[c] = (__bf16)(((xr[c] - mu) * rstd) * w[c] + bias[c]);
}

// ---------------------------------------------------------------------------
// Elementwise f32 -> bf16.
// ---------------------------------------------------------------------------
__global__ __launch_bounds__(256)
void mca_cvt_kernel(const float* __restrict__ s, __bf16* __restrict__ d, int n)
{
    int i = blockIdx.x * 256 + threadIdx.x;
    if (i < n) d[i] = (__bf16)s[i];
}

// ---------------------------------------------------------------------------
// text_time = cumsum(media_locations) per batch row (tiny serial scan).
// ---------------------------------------------------------------------------
__global__ void mca_cumsum_kernel(const unsigned char* __restrict__ loc,
                                  int* __restrict__ tt)
{
    const int b = blockIdx.x;
    if (threadIdx.x == 0) {
        int s = 0;
        for (int i = 0; i < T_TXT; ++i) {
            s += (loc[(size_t)b * T_TXT + i] != 0) ? 1 : 0;
            tt[(size_t)b * T_TXT + i] = s;
        }
    }
}

// ---------------------------------------------------------------------------
// Generic bf16 GEMM: C(MxN) = scale * A(MxK) @ B(KxN).
// 256 threads = 8 waves; block tile 128x64. Waves arranged 4(M) x 2(N),
// each wave owns a 32x32 sub-tile: 2 A-frags x 2 B-frags -> 4 WMMAs per
// K=32 chunk from 8 b128 LDS loads (2 loads/WMMA). B staged TRANSPOSED
// (Bt[n][k]). Next chunk prefetched (global_prefetch_b8).
// out_is_f32: 0 -> bf16 output, 1 -> f32 output. M%128==0, N%64==0, K%32==0.
// ---------------------------------------------------------------------------
__global__ __launch_bounds__(256)
void mca_gemm_bf16_kernel(const __bf16* __restrict__ A,
                          const __bf16* __restrict__ B,
                          void* __restrict__ Cout,
                          int M, int N, int K, float scale, int out_is_f32)
{
    __shared__ __align__(16) __bf16 As[128 * ASTR];    // [m][k]
    __shared__ __align__(16) __bf16 Bt[64 * BSTR];     // [n][k] (transposed)

    const int tid  = threadIdx.x;
    const int wave = tid >> 5;
    const int lane = tid & 31;
    const int half = lane >> 4;
    const int l16  = lane & 15;
    const int wm   = wave & 3;   // 4 waves along M (32 rows each)
    const int wn   = wave >> 2;  // 2 waves along N (32 cols each)
    const int row0 = blockIdx.x * 128;
    const int col0 = blockIdx.y * 64;

    // global->LDS load assignments
    // A: 128x32 = 4096 bf16; 2 x 16B per thread
    const int ar0 = (tid * 2) >> 2;            // rows for the 2 octet loads
    const int ac0 = ((tid * 2) & 3) << 3;
    const int ar1 = (tid * 2 + 1) >> 2;
    const int ac1 = ((tid * 2 + 1) & 3) << 3;
    // B: 32x64 = 2048 bf16; 8 per thread (16B run, scattered transpose store)
    const int br = tid >> 3;            // 0..31, B k-row
    const int bc = (tid & 7) << 3;      // 0..56, n octet

    v8f acc00 = {}, acc01 = {}, acc10 = {}, acc11 = {};

    for (int kc = 0; kc < K; kc += 32) {
        __syncthreads();
        const __bf16* ap0 = A + (size_t)(row0 + ar0) * K + kc + ac0;
        const __bf16* ap1 = A + (size_t)(row0 + ar1) * K + kc + ac1;
        *(uint4*)(&As[ar0 * ASTR + ac0]) = *(const uint4*)ap0;
        *(uint4*)(&As[ar1 * ASTR + ac1]) = *(const uint4*)ap1;
        const __bf16* bp = B + (size_t)(kc + br) * N + col0 + bc;
        v8bf bv = *(const v8bf*)bp;
        #pragma unroll
        for (int j = 0; j < 8; ++j)
            Bt[(bc + j) * BSTR + br] = bv[j];
        if (kc + 32 < K) {
            __builtin_prefetch(ap0 + 32, 0, 1);
            __builtin_prefetch(bp + (size_t)32 * N, 0, 1);
        }
        __syncthreads();

        v16bf a0 = load_afrag(&As[(wm * 32 + l16) * ASTR], half);
        v16bf a1 = load_afrag(&As[(wm * 32 + 16 + l16) * ASTR], half);
        v16bf b0 = load_bfrag(&Bt[(wn * 32 + l16) * BSTR], half);
        v16bf b1 = load_bfrag(&Bt[(wn * 32 + 16 + l16) * BSTR], half);
        acc00 = __builtin_amdgcn_wmma_f32_16x16x32_bf16(false, a0, false, b0,
                                                        (short)0, acc00, false, false);
        acc01 = __builtin_amdgcn_wmma_f32_16x16x32_bf16(false, a0, false, b1,
                                                        (short)0, acc01, false, false);
        acc10 = __builtin_amdgcn_wmma_f32_16x16x32_bf16(false, a1, false, b0,
                                                        (short)0, acc10, false, false);
        acc11 = __builtin_amdgcn_wmma_f32_16x16x32_bf16(false, a1, false, b1,
                                                        (short)0, acc11, false, false);
    }

    // C layout per tile: lanes 0-15 -> rows 0..7, lanes 16-31 -> rows 8..15
    const int crow = row0 + wm * 32 + (half << 3);
    const int ccol = col0 + wn * 32 + l16;
    if (out_is_f32) {
        float* C = (float*)Cout;
        #pragma unroll
        for (int v = 0; v < 8; ++v) {
            C[(size_t)(crow + v) * N + ccol]           = acc00[v] * scale;
            C[(size_t)(crow + v) * N + ccol + 16]      = acc01[v] * scale;
            C[(size_t)(crow + 16 + v) * N + ccol]      = acc10[v] * scale;
            C[(size_t)(crow + 16 + v) * N + ccol + 16] = acc11[v] * scale;
        }
    } else {
        __bf16* C = (__bf16*)Cout;
        #pragma unroll
        for (int v = 0; v < 8; ++v) {
            C[(size_t)(crow + v) * N + ccol]           = (__bf16)(acc00[v] * scale);
            C[(size_t)(crow + v) * N + ccol + 16]      = (__bf16)(acc01[v] * scale);
            C[(size_t)(crow + 16 + v) * N + ccol]      = (__bf16)(acc10[v] * scale);
            C[(size_t)(crow + 16 + v) * N + ccol + 16] = (__bf16)(acc11[v] * scale);
        }
    }
}

// ---------------------------------------------------------------------------
// Masked flash attention. One wave handles 16 query rows of one (b,h);
// the 8 waves of a block share (b,h) and cooperatively stage V transposed.
// 32 keys per iteration (2 sim WMMAs per 16-key tile + 4 PV WMMAs, full K=32).
// Block-diagonal mask exploited: chunks no row matches are skipped (~16x).
// q pre-scaled by DIM_HEAD^-0.5. kv packed (b, key, 2*INNER): k [0,512),
// v [512,1024). Key j allowed iff (j>>6)+1 == text_time; tt==0 rows -> 0.
// ---------------------------------------------------------------------------
__global__ __launch_bounds__(256)
void mca_attn_kernel(const __bf16* __restrict__ q, const __bf16* __restrict__ kv,
                     const int* __restrict__ tt, __bf16* __restrict__ out)
{
    __shared__ __align__(16) __bf16 vT[DIM_HEAD * VSTR];   // [d][key%32] chunk
    __shared__ __align__(16) __bf16 p_s[8][16 * PSTR];     // per-wave probs
    __shared__ int tt_s[8][16];
    __shared__ int need;

    const int tid  = threadIdx.x;
    const int wave = tid >> 5;
    const int lane = tid & 31;
    const int half = lane >> 4;
    const int l16  = lane & 15;
    const int h    = blockIdx.y;
    const int b    = blockIdx.z;
    const int i0   = (blockIdx.x * 8 + wave) * 16;

    const __bf16* qb  = q  + (size_t)b * T_TXT * INNER;
    const __bf16* kvb = kv + (size_t)b * NKEY * (2 * INNER);

    // Q fragments: two 32-wide d chunks (contiguous per-lane b128 loads)
    const __bf16* qr = qb + (size_t)(i0 + l16) * INNER + h * DIM_HEAD;
    v16bf aq0 = load_afrag(qr, half);
    v16bf aq1 = load_afrag(qr + 32, half);

    if (half == 0) tt_s[wave][l16] = tt[(size_t)b * T_TXT + i0 + l16];
    asm volatile("s_wait_dscnt 0x0" ::: "memory");

    float mrow[8], lrow[8];
    #pragma unroll
    for (int v = 0; v < 8; ++v) { mrow[v] = -3.0e38f; lrow[v] = 0.f; }
    v8f acc[4] = {};

    // V staging assignment: key = tid>>3 (0..31), d octet = (tid&7)*8
    const int skey = tid >> 3;
    const int sd0  = (tid & 7) << 3;

    for (int kt = 0; kt < NKEY / 32; ++kt) {
        const int mt = kt / 2 + 1;            // media_time of this 32-key chunk
        int act = 0;
        #pragma unroll
        for (int v = 0; v < 8; ++v) act |= (tt_s[wave][(half << 3) + v] == mt);
        const int waveActive = __any(act);

        if (tid == 0) need = 0;
        __syncthreads();
        if (waveActive && lane == 0) need = 1;
        __syncthreads();

        if (need) {   // stage V chunk transposed: vT[d][key]
            const __bf16* vp = kvb + (size_t)(kt * 32 + skey) * (2 * INNER)
                             + INNER + h * DIM_HEAD + sd0;
            v8bf vv = *(const v8bf*)vp;
            #pragma unroll
            for (int j = 0; j < 8; ++j)
                vT[(sd0 + j) * VSTR + skey] = vv[j];
        }
        __syncthreads();

        if (waveActive) {
            // --- sim tiles: keys [kt*32, +16) and [+16, +32) ---
            const __bf16* kr0 = kvb + (size_t)(kt * 32 + l16) * (2 * INNER) + h * DIM_HEAD;
            const __bf16* kr1 = kr0 + (size_t)16 * (2 * INNER);
            v8f c0 = {}, c1 = {};
            c0 = __builtin_amdgcn_wmma_f32_16x16x32_bf16(false, aq0, false,
                     load_bfrag(kr0, half), (short)0, c0, false, false);
            c0 = __builtin_amdgcn_wmma_f32_16x16x32_bf16(false, aq1, false,
                     load_bfrag(kr0 + 32, half), (short)0, c0, false, false);
            c1 = __builtin_amdgcn_wmma_f32_16x16x32_bf16(false, aq0, false,
                     load_bfrag(kr1, half), (short)0, c1, false, false);
            c1 = __builtin_amdgcn_wmma_f32_16x16x32_bf16(false, aq1, false,
                     load_bfrag(kr1 + 32, half), (short)0, c1, false, false);

            // --- mask + online softmax over the 32 keys ---
            float p0[8], p1[8];
            #pragma unroll
            for (int v = 0; v < 8; ++v) {
                const int ttv = tt_s[wave][(half << 3) + v];
                const float s0 = (ttv == mt) ? c0[v] : -3.0e38f;
                const float s1 = (ttv == mt) ? c1[v] : -3.0e38f;
                float m = fmaxf(s0, s1);
                #pragma unroll
                for (int off = 1; off < 16; off <<= 1)
                    m = fmaxf(m, __shfl_xor(m, off, 32));
                const float mnew = fmaxf(mrow[v], m);
                const float fs = __expf(mrow[v] - mnew);
                const float e0 = __expf(s0 - mnew);
                const float e1 = __expf(s1 - mnew);
                float rs = e0 + e1;
                #pragma unroll
                for (int off = 1; off < 16; off <<= 1)
                    rs += __shfl_xor(rs, off, 32);
                lrow[v] = lrow[v] * fs + rs;
                mrow[v] = mnew;
                p0[v] = e0;
                p1[v] = e1;
                #pragma unroll
                for (int t = 0; t < 4; ++t) acc[t][v] *= fs;
            }

            // --- stage probabilities row-major: p_s[row][key 0..31] ---
            asm volatile("s_wait_dscnt 0x0" ::: "memory");   // WAR vs prior reads
            #pragma unroll
            for (int v = 0; v < 8; ++v) {
                const int r = (half << 3) + v;
                p_s[wave][r * PSTR + l16]      = (__bf16)p0[v];
                p_s[wave][r * PSTR + 16 + l16] = (__bf16)p1[v];
            }
            asm volatile("s_wait_dscnt 0x0" ::: "memory");   // RAW before repack

            // --- PV: acc[t] += p(16x32) @ vT-chunk(32x16), t over d tiles ---
            v16bf ap = load_afrag(&p_s[wave][l16 * PSTR], half);
            #pragma unroll
            for (int t = 0; t < 4; ++t) {
                v16bf bv = load_bfrag(&vT[(t * 16 + l16) * VSTR], half);
                acc[t] = __builtin_amdgcn_wmma_f32_16x16x32_bf16(false, ap, false, bv,
                             (short)0, acc[t], false, false);
            }
        }
        __syncthreads();   // protect vT before next chunk's staging
    }

    // --- normalize; zero rows with text_time == 0 (or no mass); store ---
    float inv[8];
    #pragma unroll
    for (int v = 0; v < 8; ++v) {
        const int ttv = tt_s[wave][(half << 3) + v];
        inv[v] = (ttv != 0 && lrow[v] > 0.f) ? (1.0f / lrow[v]) : 0.0f;
    }
    __bf16* ob = out + ((size_t)b * T_TXT + i0 + (half << 3)) * INNER + h * DIM_HEAD;
    #pragma unroll
    for (int t = 0; t < 4; ++t)
        #pragma unroll
        for (int v = 0; v < 8; ++v)
            ob[(size_t)v * INNER + t * 16 + l16] = (__bf16)(acc[t][v] * inv[v]);
}

// ---------------------------------------------------------------------------
// Host-side launcher.
// Inputs: 0:x 1:media 2:media_locations 3:ln_w 4:ln_b 5:Wq 6:Wkv 7:Wo
// ---------------------------------------------------------------------------
extern "C" void kernel_launch(void* const* d_in, const int* in_sizes, int n_in,
                              void* d_out, int out_size, void* d_ws, size_t ws_size,
                              hipStream_t stream)
{
    const float* x    = (const float*)d_in[0];
    const float* media= (const float*)d_in[1];
    const unsigned char* mloc = (const unsigned char*)d_in[2];
    const float* ln_w = (const float*)d_in[3];
    const float* ln_b = (const float*)d_in[4];
    const float* Wq   = (const float*)d_in[5];
    const float* Wkv  = (const float*)d_in[6];
    const float* Wo   = (const float*)d_in[7];
    float* out = (float*)d_out;

    char* ws = (char*)d_ws;
    size_t off = 0;
    auto walloc = [&](size_t bytes) -> void* {
        void* p = ws + off;
        off += (bytes + 255) & ~(size_t)255;
        return p;
    };
    __bf16* xnb    = (__bf16*)walloc((size_t)M_TXT * DIMX * 2);        // LN(x) bf16
    __bf16* Wqb    = (__bf16*)walloc((size_t)DIMX * INNER * 2);
    __bf16* Wkvb   = (__bf16*)walloc((size_t)DIM_VIS * 2 * INNER * 2);
    __bf16* Wob    = (__bf16*)walloc((size_t)INNER * DIMX * 2);
    __bf16* mediab = (__bf16*)walloc((size_t)M_KV * DIM_VIS * 2);
    __bf16* qb     = (__bf16*)walloc((size_t)M_TXT * INNER * 2);
    __bf16* kvb    = (__bf16*)walloc((size_t)M_KV * 2 * INNER * 2);
    __bf16* aob    = (__bf16*)walloc((size_t)M_TXT * INNER * 2);
    int*    ttime  = (int*)   walloc((size_t)BATCH * T_TXT * 4);

    // 1) LayerNorm + bf16 conversions
    mca_ln_kernel<<<dim3(M_TXT), dim3(256), 0, stream>>>(x, ln_w, ln_b, xnb);
    mca_cvt_kernel<<<dim3((DIMX * INNER) / 256), dim3(256), 0, stream>>>(Wq, Wqb, DIMX * INNER);
    mca_cvt_kernel<<<dim3((DIM_VIS * 2 * INNER) / 256), dim3(256), 0, stream>>>(Wkv, Wkvb, DIM_VIS * 2 * INNER);
    mca_cvt_kernel<<<dim3((INNER * DIMX) / 256), dim3(256), 0, stream>>>(Wo, Wob, INNER * DIMX);
    mca_cvt_kernel<<<dim3((M_KV * DIM_VIS) / 256), dim3(256), 0, stream>>>(media, mediab, M_KV * DIM_VIS);
    mca_cumsum_kernel<<<dim3(BATCH), dim3(64), 0, stream>>>(mloc, ttime);

    // 2) q = LN(x) @ Wq, scaled by DIM_HEAD^-0.5 (fold softmax scale here)
    mca_gemm_bf16_kernel<<<dim3(M_TXT / 128, INNER / 64), dim3(256), 0, stream>>>(
        xnb, Wqb, qb, M_TXT, INNER, DIMX, 0.125f, 0);

    // 3) kv = media @ Wkv
    mca_gemm_bf16_kernel<<<dim3(M_KV / 128, (2 * INNER) / 64), dim3(256), 0, stream>>>(
        mediab, Wkvb, kvb, M_KV, 2 * INNER, DIM_VIS, 1.0f, 0);

    // 4) masked flash attention (block-diagonal skip)
    mca_attn_kernel<<<dim3(T_TXT / 16 / 8, HEADS, BATCH), dim3(256), 0, stream>>>(
        qb, kvb, ttime, aob);

    // 5) out = attn_out @ Wo (f32 output)
    mca_gemm_bf16_kernel<<<dim3(M_TXT / 128, DIMX / 64), dim3(256), 0, stream>>>(
        aob, Wob, (void*)out, M_TXT, DIMX, INNER, 1.0f, 1);
}
